// ModDemodConv3x3_71871982731630
// MI455X (gfx1250) — matrix operand order repaired and checked by reading, not compiled, not run on hardware
//
#include <hip/hip_runtime.h>

// ---------------------------------------------------------------------------
// ModDemodConv3x3 for MI455X (gfx1250):  out = demod(style * W) (x) x
// B=8, CIN=COUT=256, H=W=64, 3x3, pad 1.
// 9 shifted 1x1 GEMMs on v_wmma_f32_16x16x32_bf16 (f32 accum), with
// Tensor-Data-Mover (tensor_load_to_lds) double-buffered LDS staging.
// ---------------------------------------------------------------------------

#define BSZ   8
#define CIN   256
#define COUT  256
#define HH    64
#define WW    64
#define TAPS  9
#define KCHUNKS (CIN / 32)            // 8 chunks of K=32 per tap
#define CPAD  40                      // channel-dim padding (80B col stride)
#define COLS  66                      // input cols -1..64  (halo baked in)
#define ROWSP 66                      // input rows -1..64  (halo baked in)
#define TILE_SHORTS (3 * COLS * CPAD) // 7920 bf16 = 15840 B per K-chunk tile
#define ROW_SHORTS  (COLS * CPAD)     // 2640 bf16 per padded input row
#define PACK_PER_BLK (TAPS * KCHUNKS * 32 * 16)   // 36864 bf16 per (b, mtile)
#define DWPACK_SHORTS ((size_t)BSZ * 16 * PACK_PER_BLK)   // 4.72M bf16

typedef __attribute__((ext_vector_type(16))) __bf16 bf16x16;
typedef __attribute__((ext_vector_type(8)))  __bf16 bf16x8;
typedef __attribute__((ext_vector_type(8)))  float  f32x8;
typedef __attribute__((ext_vector_type(4)))  unsigned int uint32x4;
typedef __attribute__((ext_vector_type(8)))  int  int32x8;
typedef __attribute__((ext_vector_type(4)))  int  int32x4;

__device__ __forceinline__ unsigned short f32_to_bf16(float f) {
    union { float f; unsigned u; } v; v.f = f;
    unsigned u = v.u;
    u += 0x7FFFu + ((u >> 16) & 1u);      // round-to-nearest-even
    return (unsigned short)(u >> 16);
}

// ---------------------------------------------------------------------------
// Kernel 1: modulate + demodulate weights, write bf16 in A-fragment order.
// dwpack layout: [b][mtile(16)][tap(9)][kchunk(8)][lane(32)][j(16)]  bf16
//   lane L: M-row = L%16, hi = L/16
//   element j<8 :  K = hi*8 + j ;  j>=8:  K = 16 + hi*8 + (j-8)
// ---------------------------------------------------------------------------
__global__ void __launch_bounds__(256)
moddemod_pack_kernel(const float* __restrict__ style,
                     const float* __restrict__ weight,
                     unsigned short* __restrict__ dwpack) {
    const int blk   = blockIdx.x;          // b*16 + mtile
    const int b     = blk >> 4;
    const int mtile = blk & 15;
    const int tid   = threadIdx.x;

    __shared__ float red[256];
    __shared__ float rsig[16];

    const int olocal = tid >> 4;
    const int isub   = tid & 15;
    const int o      = mtile * 16 + olocal;

    float partial = 0.f;
    for (int i = isub; i < CIN; i += 16) {
        const float s = style[b * CIN + i];
        const float* wp = &weight[(size_t)(o * CIN + i) * TAPS];
        float wsum = 0.f;
#pragma unroll
        for (int t = 0; t < TAPS; ++t) { const float w = wp[t]; wsum += w * w; }
        partial += s * s * wsum;
    }
    red[tid] = partial;
    __syncthreads();
    if (isub == 0) {
        float s = 0.f;
#pragma unroll
        for (int k = 0; k < 16; ++k) s += red[olocal * 16 + k];
        rsig[olocal] = 1.0f / sqrtf(s + 1e-8f);
    }
    __syncthreads();

    const size_t base = (size_t)blk * PACK_PER_BLK;
    for (int idx = tid; idx < PACK_PER_BLK; idx += 256) {
        const int tap   = idx >> 12;
        const int rem   = idx & 4095;
        const int chunk = rem >> 9;
        const int rem2  = rem & 511;
        const int lane  = rem2 >> 4;
        const int j     = rem2 & 15;
        const int hi    = lane >> 4;
        const int m     = lane & 15;
        const int K     = (j < 8) ? (hi * 8 + j) : (16 + hi * 8 + (j - 8));
        const int i     = chunk * 32 + K;
        const int oo    = mtile * 16 + m;
        const float v = style[b * CIN + i]
                      * weight[(size_t)(oo * CIN + i) * TAPS + tap]
                      * rsig[m];
        dwpack[base + idx] = f32_to_bf16(v);
    }
}

// ---------------------------------------------------------------------------
// Kernel 2: pre-pack x into TDM-friendly bf16 layout with halo + padding:
//   xT[b][chunk(8)][hin_p(66)][col(66)][CPAD(40)]  bf16
// hin_p = input row + 1 (rows -1 and 64 are zero), col = input col + 1.
// Channels innermost (transposed) so B fragments are 2x ds_load_b128,
// and each conv K-chunk tile is one contiguous 15840B block for the TDM.
// ---------------------------------------------------------------------------
__global__ void __launch_bounds__(256)
prepack_x_kernel(const float* __restrict__ x,
                 unsigned short* __restrict__ xT) {
    const int blk   = blockIdx.x;                  // b*8*66 + chunk*66 + hin_p
    const int hin_p = blk % ROWSP;
    const int chunk = (blk / ROWSP) & 7;
    const int b     = blk / (ROWSP * 8);
    const int tid   = threadIdx.x;
    const int hin   = hin_p - 1;

    __shared__ unsigned short tile[ROW_SHORTS];

    for (int idx = tid; idx < ROW_SHORTS; idx += 256) tile[idx] = 0;
    __syncthreads();

    if ((unsigned)hin < (unsigned)HH) {
        // coalesced reads: w fastest; transposed LDS write [col][c]
        for (int idx = tid; idx < 32 * WW; idx += 256) {
            const int c = idx >> 6;
            const int w = idx & 63;
            const float v =
                x[(((size_t)b * CIN + chunk * 32 + c) * HH + hin) * WW + w];
            tile[(w + 1) * CPAD + c] = f32_to_bf16(v);
        }
    }
    __syncthreads();

    unsigned short* dst = xT + (size_t)blk * ROW_SHORTS;  // fully contiguous
    for (int idx = tid; idx < ROW_SHORTS; idx += 256) dst[idx] = tile[idx];
}

// ---------------------------------------------------------------------------
// TDM issue: 1-D copy of one K-chunk tile (3 rows x 66 cols x 40 ch bf16 =
// 3960 dwords) from xT into LDS.  D# built per CDNA5 ISA bitfields.
// ---------------------------------------------------------------------------
__device__ __forceinline__ void tdm_load_tile(const unsigned short* src,
                                              const unsigned short* lds_dst) {
    const unsigned long long ga = (unsigned long long)(const void*)src;
    const unsigned lds_addr = (unsigned)(unsigned long long)(const void*)lds_dst;
    const unsigned NDW = TILE_SHORTS / 2;         // 3960 dwords

    uint32x4 g0;
    g0[0] = 1u;                                   // count=1 valid descriptor
    g0[1] = lds_addr;                             // lds_addr [63:32]
    g0[2] = (unsigned)ga;                         // global_addr [95:64]
    g0[3] = (unsigned)(ga >> 32) | 0x80000000u;   // addr[56:32] | type=2

    int32x8 g1;
    g1[0] = (int)(2u << 16);                      // data_size=4B, mask=0
    g1[1] = (int)((NDW & 0xFFFFu) << 16);         // tensor_dim0 lo16
    g1[2] = (int)(((NDW >> 16) & 0xFFFFu) | (1u << 16)); // dim0 hi | dim1=1
    g1[3] = (int)(NDW << 16);                     // tile_dim0 = 3960
    g1[4] = 0;                                    // tile_dim1=0 (1-D), dim2=0
    g1[5] = (int)NDW;                             // tensor_dim0_stride lo32
    g1[6] = 0;
    g1[7] = 0;

    const int32x4 z4 = {0, 0, 0, 0};
#if __clang_major__ >= 23
    const int32x8 z8 = {0, 0, 0, 0, 0, 0, 0, 0};
    __builtin_amdgcn_tensor_load_to_lds(g0, g1, z4, z4, z8, 0);
#else
    __builtin_amdgcn_tensor_load_to_lds(g0, g1, z4, z4, 0);
#endif
}

// ---------------------------------------------------------------------------
// Kernel 3: implicit GEMM conv. One block per (b, output row h).
// 8 waves; wave wv owns mtiles {2wv,2wv+1} x 4 pixel tiles (32x64 out tile).
// TDM double-buffers the x tile; 72 WMMAs per K-chunk per wave.
// ---------------------------------------------------------------------------
__global__ void __launch_bounds__(256)
modconv_wmma_kernel(const unsigned short* __restrict__ xT,
                    const unsigned short* __restrict__ dwpack,
                    float* __restrict__ out) {
    const int h    = blockIdx.x;
    const int b    = blockIdx.y;
    const int tid  = threadIdx.x;
    const int wv   = tid >> 5;
    const int lane = tid & 31;
    const int n    = lane & 15;
    const int hi   = lane >> 4;

    __shared__ __align__(16) unsigned short xs[2][TILE_SHORTS];  // 31680 B

    f32x8 acc[2][4];
    const f32x8 zero = {};
#pragma unroll
    for (int mi = 0; mi < 2; ++mi)
#pragma unroll
        for (int nt = 0; nt < 4; ++nt) acc[mi][nt] = zero;

    const unsigned short* wb = dwpack + (size_t)b * 16 * PACK_PER_BLK;
    // chunk tile = rows h..h+2 (padded coords) of xT[b][chunk]
    const unsigned short* xb =
        xT + ((size_t)(b * 8) * ROWSP + h) * ROW_SHORTS;

    if (wv == 0) tdm_load_tile(xb, &xs[0][0]);      // prefetch chunk 0

#pragma unroll 1
    for (int chunk = 0; chunk < KCHUNKS; ++chunk) {
        if (wv == 0) __builtin_amdgcn_s_wait_tensorcnt(0);
        __syncthreads();                            // publish buffer chunk&1
        if (chunk + 1 < KCHUNKS && wv == 0)
            tdm_load_tile(xb + (size_t)(chunk + 1) * ROWSP * ROW_SHORTS,
                          &xs[(chunk + 1) & 1][0]);

        const unsigned short* buf = &xs[chunk & 1][0];

#pragma unroll
        for (int tap = 0; tap < TAPS; ++tap) {
            const int kh = tap / 3;
            const int kw = tap % 3;

            // A fragments: contiguous 32B per lane, prepacked by kernel 1
            const size_t a0off = ((size_t)(2 * wv) * (TAPS * KCHUNKS)
                                  + tap * KCHUNKS + chunk) * 512 + lane * 16;
            const size_t a1off = ((size_t)(2 * wv + 1) * (TAPS * KCHUNKS)
                                  + tap * KCHUNKS + chunk) * 512 + lane * 16;
            const bf16x16 a0 = *(const bf16x16*)(const void*)(wb + a0off);
            const bf16x16 a1 = *(const bf16x16*)(const void*)(wb + a1off);

#pragma unroll
            for (int nt = 0; nt < 4; ++nt) {
                // B fragment: j<8: K = hi*8 + j; j>=8: K = 16 + hi*8 + (j-8),
                // pixel col = nt*16 + n shifted by kw; 2x aligned b128.
                const int colIdx = nt * 16 + n + kw;           // 0..65
                const unsigned short* bp =
                    buf + (size_t)(kh * COLS + colIdx) * CPAD + hi * 8;
                union { bf16x16 v; struct { bf16x8 lo, hi; } p; } bf;
                bf.p.lo = *(const bf16x8*)(const void*)(bp);
                bf.p.hi = *(const bf16x8*)(const void*)(bp + 16);

                acc[0][nt] = __builtin_amdgcn_wmma_f32_16x16x32_bf16(
                    false, a0, false, bf.v, (short)0, acc[0][nt], false, false);
                acc[1][nt] = __builtin_amdgcn_wmma_f32_16x16x32_bf16(
                    false, a1, false, bf.v, (short)0, acc[1][nt], false, false);
            }
        }
    }

    // ---- epilogue: C/D layout (VGPR r: M=r+8*hi, N=lane%16) ----
#pragma unroll
    for (int mi = 0; mi < 2; ++mi) {
        const int o_base = (2 * wv + mi) * 16 + hi * 8;
#pragma unroll
        for (int nt = 0; nt < 4; ++nt) {
            const int w = nt * 16 + n;
#pragma unroll
            for (int r = 0; r < 8; ++r) {
                const int o = o_base + r;
                out[(((size_t)b * COUT + o) * HH + h) * WW + w] = acc[mi][nt][r];
            }
        }
    }
}

// ---------------------------------------------------------------------------
extern "C" void kernel_launch(void* const* d_in, const int* in_sizes, int n_in,
                              void* d_out, int out_size, void* d_ws, size_t ws_size,
                              hipStream_t stream) {
    const float* x      = (const float*)d_in[0];   // [8,256,64,64]
    const float* style  = (const float*)d_in[1];   // [8,256]
    const float* weight = (const float*)d_in[2];   // [256,256,3,3]
    float* out          = (float*)d_out;           // [8,256,64,64]

    unsigned short* dwpack = (unsigned short*)d_ws;              // 9.44 MB
    unsigned short* xT     = dwpack + DWPACK_SHORTS;             // 22.3 MB

    (void)in_sizes; (void)n_in; (void)out_size; (void)ws_size;

    moddemod_pack_kernel<<<BSZ * 16, 256, 0, stream>>>(style, weight, dwpack);
    prepack_x_kernel<<<BSZ * 8 * ROWSP, 256, 0, stream>>>(x, xT);
    modconv_wmma_kernel<<<dim3(HH, BSZ), 256, 0, stream>>>(xT, dwpack, out);
}
